// Gaussian_Upsampler_50294067036848
// MI455X (gfx1250) — compile-verified, matrix operand order checked by reading, not run on hardware
//
#include <hip/hip_runtime.h>
#include <math.h>

// ---------------- problem constants (from reference setup) ----------------
#define BB   32      // batch
#define CC   256     // channels / lstm input width (= 2*H)
#define TT   512     // encoder time steps
#define HH   128     // lstm hidden
#define GG   512     // 4*H gates
#define KW   5       // conv kernel width

#define EPSF      1e-5f
#define HALF_L2PI 0.9189385332046727f   // 0.5*log(2*pi)

typedef __attribute__((ext_vector_type(16))) __bf16 v16bf;
typedef __attribute__((ext_vector_type(8)))  float  v8f;

// ---------------------------------------------------------------------------
// WMMA fragment loader: row-major source, tile origin `tile`, leading dim ld.
// A-matrix 16x32 bf16 layout (ISA 7.12.2): lane l (row l&15) holds
//   K = kh..kh+7 (v0..v3) and K = kh+16..kh+23 (v4..v7), kh = 8*(l>=16).
// For B = W^T with W row-major (NxK), the identical gather over W rows works.
// ---------------------------------------------------------------------------
__device__ __forceinline__ v16bf load_frag_rm(const __bf16* tile, int ld, int lane) {
  const int r  = lane & 15;
  const int kh = (lane >> 4) << 3;   // 0 or 8
  const __bf16* p = tile + r * ld + kh;
  v16bf v;
#pragma unroll
  for (int i = 0; i < 8; ++i) v[i] = p[i];
#pragma unroll
  for (int i = 0; i < 8; ++i) v[i + 8] = p[16 + i];
  return v;
}

__device__ __forceinline__ float sigmoidf_(float x) {
  return 1.0f / (1.0f + __expf(-x));
}

// async global -> LDS 128-bit copy (CDNA5, tracked by ASYNCcnt).
// lds_off: wave-relative LDS byte address (low 32 bits of a generic
// __shared__ pointer); gptr: per-lane global address.
__device__ __forceinline__ void async_copy_b128(uint32_t lds_off, const void* gptr) {
  asm volatile("global_load_async_to_lds_b128 %0, %1, off"
               :: "v"(lds_off), "v"(gptr)
               : "memory");
}

// ---------------------------------------------------------------------------
// 1) conv1d(durations) + encodings  ->  x[t][b][c]   (t,b,c layout for GEMM)
// ---------------------------------------------------------------------------
__global__ void conv_add_kernel(const float* __restrict__ enc,
                                const int*   __restrict__ dur,
                                const float* __restrict__ cw,
                                const float* __restrict__ cb,
                                float* __restrict__ x) {
  int id = blockIdx.x * blockDim.x + threadIdx.x;   // ((t*B)+b)*C + c
  if (id >= TT * BB * CC) return;
  int c  = id & (CC - 1);
  int tb = id >> 8;
  int b  = tb & (BB - 1);
  int t  = tb >> 5;
  float acc = cb[c];
#pragma unroll
  for (int k = 0; k < KW; ++k) {
    int tt = t + k - (KW - 1) / 2;
    if (tt >= 0 && tt < TT) acc += cw[c * KW + k] * (float)dur[b * TT + tt];
  }
  x[id] = enc[(b * CC + c) * TT + t] + acc;
}

// ---------------------------------------------------------------------------
// small helpers: f32 -> bf16 cast, bias sum
// ---------------------------------------------------------------------------
__global__ void cast_bf16_kernel(const float* __restrict__ s, __bf16* __restrict__ d, int n) {
  int i = blockIdx.x * blockDim.x + threadIdx.x;
  if (i < n) d[i] = (__bf16)s[i];
}
__global__ void bias_sum_kernel(const float* __restrict__ a, const float* __restrict__ b,
                                float* __restrict__ d, int n) {
  int i = blockIdx.x * blockDim.x + threadIdx.x;
  if (i < n) d[i] = a[i] + b[i];
}

// ---------------------------------------------------------------------------
// 2) input projection GEMM:  xg[d][t][b][g] = X[t][b][:] . Wih[d][g][:] + bias
//    M = T*B = 16384, K = C = 256, N = G = 512.  One wave per 16x16 tile.
// ---------------------------------------------------------------------------
__global__ void __launch_bounds__(32)
gemm_xg_kernel(const __bf16* __restrict__ X,
               const __bf16* __restrict__ Wih,   // [2][G][C]
               const float*  __restrict__ bias,  // [2][G]
               float* __restrict__ xg) {         // [2][T*B][G]
  const int dir    = blockIdx.y;
  const int Ntiles = GG / 16;                    // 32
  const int mtile  = blockIdx.x / Ntiles;
  const int ntile  = blockIdx.x % Ntiles;
  const int lane   = threadIdx.x;

  const __bf16* Wd  = Wih + (size_t)dir * GG * CC;
  const float*  bd  = bias + dir * GG;
  float*        od  = xg + (size_t)dir * TT * BB * GG;

  const int n = ntile * 16 + (lane & 15);
  v8f acc;
  const float bv = bd[n];
#pragma unroll
  for (int i = 0; i < 8; ++i) acc[i] = bv;

  const __bf16* Xt = X  + (size_t)mtile * 16 * CC;
  const __bf16* Wt = Wd + (size_t)ntile * 16 * CC;
#pragma unroll
  for (int k0 = 0; k0 < CC; k0 += 32) {
    v16bf a = load_frag_rm(Xt + k0, CC, lane);
    v16bf b = load_frag_rm(Wt + k0, CC, lane);
    acc = __builtin_amdgcn_wmma_f32_16x16x32_bf16(false, a, false, b,
                                                  (short)0, acc, false, false);
  }
  const int mbase = mtile * 16 + ((lane >> 4) << 3);
#pragma unroll
  for (int r = 0; r < 8; ++r) od[(size_t)(mbase + r) * GG + n] = acc[r];
}

// ---------------------------------------------------------------------------
// 3) LSTM recurrence (persistent workgroup per direction).
//    CDNA5 LDS budget (320KB/WGP): W_hh 128KB + h 8KB + 2x64KB async-staged
//    xg double buffer = 264KB resident.  Step s+1's 64KB of gate
//    pre-activations is fetched with GLOBAL_LOAD_ASYNC_TO_LDS_B128 while the
//    WMMAs of step s run; s_wait_asynccnt + workgroup barrier publish it.
//    16 waves: wave w -> (mtile = w&1, htile = w>>1); each wave computes the
//    i/f/g/o gate tiles for the SAME 16 hidden columns, so the cell update is
//    pure per-lane register math on matched 16x16 f32 accumulator layouts.
// ---------------------------------------------------------------------------
__global__ void __launch_bounds__(512)
lstm_recur_kernel(const float*  __restrict__ xg,    // [2][T*B][G]
                  const __bf16* __restrict__ Whh,   // [2][G][H]
                  float* __restrict__ out) {        // [T][B][2H], write cols dir*H..
  extern __shared__ char smem_raw[];
  __bf16* sW   = (__bf16*)smem_raw;                          // G*H bf16  (128 KB)
  __bf16* sh   = (__bf16*)(smem_raw + GG * HH * 2);          // B*H bf16  (  8 KB)
  float*  sxg0 = (float*) (smem_raw + GG * HH * 2 + BB * HH * 2);        // 64 KB
  float*  sxg1 = (float*) (smem_raw + GG * HH * 2 + BB * HH * 2 + BB * GG * 4);

  const int dir   = blockIdx.x;
  const int tid   = threadIdx.x;
  const int lane  = tid & 31;
  const int wave  = tid >> 5;
  const int mtile = wave & 1;        // batch tile (0..1)
  const int htile = wave >> 1;       // hidden tile (0..7)

  const float*  xgd = xg  + (size_t)dir * TT * BB * GG;
  const __bf16* Wd  = Whh + (size_t)dir * GG * HH;

  // stage W_hh into LDS (coalesced 32-bit copies); zero h0
  {
    const uint32_t* s = (const uint32_t*)Wd;
    uint32_t*       d = (uint32_t*)sW;
    for (int i = tid; i < GG * HH / 2; i += 512) d[i] = s[i];
    uint32_t* z = (uint32_t*)sh;
    for (int i = tid; i < BB * HH / 2; i += 512) z[i] = 0u;   // h0 = 0
  }
  v8f c = {};                        // cell state, registers, whole T loop

  // wave-relative LDS byte offsets of the xg staging buffers
  const uint32_t sxg_off[2] = { (uint32_t)(uintptr_t)sxg0,
                                (uint32_t)(uintptr_t)sxg1 };
  float* const sxg_ptr[2] = { sxg0, sxg1 };

  // prologue: async-stage step 0's 64KB gate tile into buffer 0
  {
    const char* g0 = (const char*)(xgd + (size_t)(dir ? (TT - 1) : 0) * BB * GG);
#pragma unroll
    for (int i = 0; i < 8; ++i) {
      const uint32_t ofs = (uint32_t)(i * 512 + tid) * 16u;
      async_copy_b128(sxg_off[0] + ofs, g0 + ofs);
    }
  }
  __syncthreads();

  const int nloc  = lane & 15;
  const int rbase = (lane >> 4) << 3;
  const int hcol  = htile * 16 + nloc;          // hidden index 0..127

  for (int s = 0; s < TT; ++s) {
    const int t   = dir ? (TT - 1 - s) : s;
    const int cur = s & 1, nxt = cur ^ 1;

    // issue async staging of step s+1 (overlaps this step's WMMAs),
    // then wait for step s's batch (8 in-flight ops -> previous 8 are done;
    // async loads complete in order).
    if (s + 1 < TT) {
      const int tn = dir ? (TT - 2 - s) : (s + 1);
      const char* gn = (const char*)(xgd + (size_t)tn * BB * GG);
#pragma unroll
      for (int i = 0; i < 8; ++i) {
        const uint32_t ofs = (uint32_t)(i * 512 + tid) * 16u;
        async_copy_b128(sxg_off[nxt] + ofs, gn + ofs);
      }
      asm volatile("s_wait_asynccnt 8" ::: "memory");
    } else {
      asm volatile("s_wait_asynccnt 0" ::: "memory");
    }
    __syncthreads();                            // staged xg + sh visible to all

    // A fragments of h (16 batch rows x K=128, 4 k-slices)
    v16bf a[4];
#pragma unroll
    for (int ks = 0; ks < 4; ++ks)
      a[ks] = load_frag_rm(sh + mtile * 16 * HH + ks * 32, HH, lane);

    const float* xgt = sxg_ptr[cur];
    v8f g[4];
#pragma unroll
    for (int gi = 0; gi < 4; ++gi) {            // i, f, g, o gate tiles
      const int n = gi * HH + hcol;
      v8f acc;
#pragma unroll
      for (int r = 0; r < 8; ++r)
        acc[r] = xgt[(mtile * 16 + rbase + r) * GG + n];
      const __bf16* Wt = sW + (size_t)(gi * HH + htile * 16) * HH;
#pragma unroll
      for (int ks = 0; ks < 4; ++ks) {
        v16bf b = load_frag_rm(Wt + ks * 32, HH, lane);
        acc = __builtin_amdgcn_wmma_f32_16x16x32_bf16(false, a[ks], false, b,
                                                      (short)0, acc, false, false);
      }
      g[gi] = acc;
    }
    __syncthreads();                            // all waves done reading sh

    float* ot = out + (size_t)t * BB * (2 * HH) + dir * HH;
#pragma unroll
    for (int r = 0; r < 8; ++r) {
      const float ig = sigmoidf_(g[0][r]);
      const float fg = sigmoidf_(g[1][r]);
      const float gg = tanhf(g[2][r]);
      const float og = sigmoidf_(g[3][r]);
      const float cv = fg * c[r] + ig * gg;
      c[r] = cv;
      const float hv = og * tanhf(cv);
      const int  b   = mtile * 16 + rbase + r;
      sh[b * HH + hcol]     = (__bf16)hv;       // feed next step
      ot[b * 2 * HH + hcol] = hv;               // hs output / next layer input
    }
    __syncthreads();                            // sh updated for next step
  }
}

// ---------------------------------------------------------------------------
// 4) projection + softplus -> 1/scale and A = -log(scale) - 0.5*log(2pi)
//    (invalid t gets A = -inf so exp() -> 0, matching ali*valid)
// ---------------------------------------------------------------------------
__global__ void proj_kernel(const float* __restrict__ xf,   // [T][B][2H]
                            const float* __restrict__ pw,
                            const float* __restrict__ pb,
                            const int*   __restrict__ lens,
                            float* __restrict__ invs,       // [B][T]
                            float* __restrict__ afac) {     // [B][T]
  int g = blockIdx.x * blockDim.x + threadIdx.x;
  if (g >= TT * BB) return;
  const int t = g >> 5, b = g & (BB - 1);
  const float* row = xf + (size_t)g * (2 * HH);
  float acc = pb[0];
  for (int i = 0; i < 2 * HH; ++i) acc += row[i] * pw[i];
  const bool valid = t < lens[b];
  const float sp   = acc > 20.f ? acc : log1pf(__expf(acc));
  const float sc   = fmaxf(valid ? sp : EPSF, EPSF);
  invs[b * TT + t] = 1.0f / sc;
  afac[b * TT + t] = valid ? (-__logf(sc) - HALF_L2PI) : -INFINITY;
}

// loc[b][t] = cumsum(dur)[b][t] - 0.5*dur[b][t]
__global__ void loc_kernel(const int* __restrict__ dur, float* __restrict__ loc) {
  const int b = blockIdx.x;
  if (threadIdx.x != 0) return;
  float cum = 0.f;
  for (int t = 0; t < TT; ++t) {
    const float d = (float)dur[b * TT + t];
    cum += d;
    loc[b * TT + t] = cum - 0.5f * d;
  }
}

// ---------------------------------------------------------------------------
// 5) gaussian alignment + normalization over t.  One block per (b, j-chunk);
//    loc/1/scale/A cached in LDS; sum pass + recompute-write pass.
// ---------------------------------------------------------------------------
__global__ void __launch_bounds__(256)
ali_kernel(const float* __restrict__ loc, const float* __restrict__ invs,
           const float* __restrict__ afac, float* __restrict__ out, int dec_t) {
  __shared__ float sL[TT], sI[TT], sA[TT];
  const int b = blockIdx.x;
  for (int i = threadIdx.x; i < TT; i += 256) {
    sL[i] = loc[b * TT + i];
    sI[i] = invs[b * TT + i];
    sA[i] = afac[b * TT + i];
  }
  __syncthreads();
  const int j = blockIdx.y * 256 + threadIdx.x;
  if (j >= dec_t) return;
  const float tj = (float)(j + 1);
  float sum = 0.f;
  for (int t = 0; t < TT; ++t) {
    const float z = (tj - sL[t]) * sI[t];
    sum += __expf(fmaf(-0.5f * z, z, sA[t]));
  }
  const float inv = 1.0f / (sum + EPSF);
  float* ob = out + (size_t)b * TT * dec_t + j;
  for (int t = 0; t < TT; ++t) {
    const float z = (tj - sL[t]) * sI[t];
    ob[(size_t)t * dec_t] = __expf(fmaf(-0.5f * z, z, sA[t])) * inv;
  }
}

// ---------------------------------------------------------------------------
extern "C" void kernel_launch(void* const* d_in, const int* in_sizes, int n_in,
                              void* d_out, int out_size, void* d_ws, size_t ws_size,
                              hipStream_t stream) {
  (void)in_sizes; (void)n_in; (void)ws_size;
  const float* enc  = (const float*)d_in[0];
  const int*   lens = (const int*)  d_in[1];
  const int*   dur  = (const int*)  d_in[2];
  // d_in[3] = dec_t (device scalar) — derive on host instead (graph-safe):
  const int dec_t   = out_size / (BB * TT);
  const float* cw   = (const float*)d_in[4];
  const float* cb   = (const float*)d_in[5];
  // lstm_params flattened: layer*8 + {Wif,Whf,bif,bhf,Wib,Whb,bib,bhb} at d_in[6..21]
  const float* pw   = (const float*)d_in[22];
  const float* pb   = (const float*)d_in[23];

  // ---- workspace carve (aligned, ~89 MB) ----
  char* w = (char*)d_ws;
  auto carve = [&](size_t bytes) { char* p = w; w += (bytes + 255) & ~size_t(255); return p; };
  float*  x     = (float*) carve((size_t)TT * BB * CC * 4);        // 16 MB (reused per layer)
  __bf16* xbf   = (__bf16*)carve((size_t)TT * BB * CC * 2);        //  8 MB
  float*  xg    = (float*) carve((size_t)2 * TT * BB * GG * 4);    // 64 MB
  __bf16* wihbf = (__bf16*)carve((size_t)2 * GG * CC * 2);
  __bf16* whhbf = (__bf16*)carve((size_t)2 * GG * HH * 2);
  float*  biasf = (float*) carve((size_t)2 * GG * 4);
  float*  locb  = (float*) carve((size_t)BB * TT * 4);
  float*  invs  = (float*) carve((size_t)BB * TT * 4);
  float*  afac  = (float*) carve((size_t)BB * TT * 4);

  // 1) conv + add -> x[t][b][c]
  conv_add_kernel<<<(TT * BB * CC) / 256, 256, 0, stream>>>(enc, dur, cw, cb, x);
  // loc (independent)
  loc_kernel<<<BB, 32, 0, stream>>>(dur, locb);

  // LDS: W_hh (128KB) + h (8KB) + xg double buffer (2x64KB) = 264KB (<320KB/WGP)
  const int shbytes = GG * HH * 2 + BB * HH * 2 + 2 * BB * GG * 4;
  for (int layer = 0; layer < 2; ++layer) {
    const int base = 6 + layer * 8;
    // weight casts (fwd at offset 0, bwd at offset 1)
    cast_bf16_kernel<<<(GG * CC + 255) / 256, 256, 0, stream>>>((const float*)d_in[base + 0], wihbf,            GG * CC);
    cast_bf16_kernel<<<(GG * CC + 255) / 256, 256, 0, stream>>>((const float*)d_in[base + 4], wihbf + GG * CC,  GG * CC);
    cast_bf16_kernel<<<(GG * HH + 255) / 256, 256, 0, stream>>>((const float*)d_in[base + 1], whhbf,            GG * HH);
    cast_bf16_kernel<<<(GG * HH + 255) / 256, 256, 0, stream>>>((const float*)d_in[base + 5], whhbf + GG * HH,  GG * HH);
    bias_sum_kernel<<<(GG + 255) / 256, 256, 0, stream>>>((const float*)d_in[base + 2], (const float*)d_in[base + 3], biasf,      GG);
    bias_sum_kernel<<<(GG + 255) / 256, 256, 0, stream>>>((const float*)d_in[base + 6], (const float*)d_in[base + 7], biasf + GG, GG);
    // activations to bf16
    cast_bf16_kernel<<<(TT * BB * CC + 255) / 256, 256, 0, stream>>>(x, xbf, TT * BB * CC);
    // input GEMM (both directions): 16384x256x512, WMMA bf16
    gemm_xg_kernel<<<dim3((TT * BB / 16) * (GG / 16), 2), 32, 0, stream>>>(xbf, wihbf, biasf, xg);
    // sequential recurrence; overwrites x with [fwd || bwd] hidden states
    lstm_recur_kernel<<<2, 512, shbytes, stream>>>(xg, whhbf, x);
  }

  // 4) projection -> 1/scale, log-factor
  proj_kernel<<<(TT * BB) / 256, 256, 0, stream>>>(x, pw, pb, lens, invs, afac);

  // 5) gaussian alignment, normalized over t
  ali_kernel<<<dim3(BB, (dec_t + 255) / 256), 256, 0, stream>>>(locb, invs, afac,
                                                                (float*)d_out, dec_t);
}
